// LlamaAttention_32787780338364
// MI455X (gfx1250) — compile-verified, matrix operand order checked by reading, not run on hardware
//
#include <hip/hip_runtime.h>
#include <hip/hip_bf16.h>

typedef __bf16 bf16;
typedef __attribute__((ext_vector_type(16))) __bf16 v16bf;
typedef __attribute__((ext_vector_type(8)))  __bf16 v8bf;
typedef __attribute__((ext_vector_type(8)))  float  v8f;

#if defined(__gfx1250__)
#define HAS_ASYNC_LDS 1
#else
#define HAS_ASYNC_LDS 0
#endif

union FragU { v16bf v; v8bf h[2]; };

// Load a 16x32 (bf16) WMMA operand fragment. `rowptr` points at [row][k0] for
// this lane's row (A: M=lane%16, B: N=lane%16 since B is read column-major).
// Per ISA layout: elems 0-7 -> K = 8*half + 0..7 ; elems 8-15 -> K = 16 + 8*half + 0..7.
__device__ __forceinline__ v16bf load_frag(const bf16* rowptr, int half) {
  FragU f;
  f.h[0] = *(const v8bf*)(rowptr + 8 * half);
  f.h[1] = *(const v8bf*)(rowptr + 16 + 8 * half);
  return f.v;
}

__device__ __forceinline__ v8f wmma_bf16(v16bf a, v16bf b, v8f c) {
  return __builtin_amdgcn_wmma_f32_16x16x32_bf16(false, a, false, b, (short)0, c,
                                                 false, false);
}

// 16B global -> LDS copy. Uses CDNA5 GLOBAL_LOAD_ASYNC_TO_LDS_B128 (no VGPR
// round-trip, tracked with ASYNCcnt) via inline asm; the LDS operand is the
// low 32 bits of the generic pointer (flat->LDS aperture uses addr[31:0]).
__device__ __forceinline__ void copy16B(const bf16* g, bf16* l) {
#if HAS_ASYNC_LDS
  unsigned lds32 = (unsigned)(unsigned long long)l;
  unsigned long long ga = (unsigned long long)g;
  asm volatile("global_load_async_to_lds_b128 %0, %1, off"
               :
               : "v"(lds32), "v"(ga)
               : "memory");
#else
  *(v8bf*)l = *(const v8bf*)g;
#endif
}

// Wait for this wave's staging to land in LDS, then workgroup barrier.
__device__ __forceinline__ void stage_sync() {
#if HAS_ASYNC_LDS
  asm volatile("s_wait_asynccnt 0" ::: "memory");
#endif
  __syncthreads();
}

// ---------------------------------------------------------------------------
// f32 -> bf16 conversion
// ---------------------------------------------------------------------------
__global__ void __launch_bounds__(256) cvt_f32_bf16(const float* __restrict__ in,
                                                    bf16* __restrict__ out, int n) {
  int i = blockIdx.x * blockDim.x + threadIdx.x;
  int stride = gridDim.x * blockDim.x;
  for (; i < n; i += stride) out[i] = (bf16)in[i];
}

// ---------------------------------------------------------------------------
// Tiled WMMA GEMM core: C[128x128 block] = A(MxK) * W(NxK)^T
// 256 threads = 8 waves arranged 4 (rows of 32) x 2 (cols of 64).
// K-step = 64 -> 16 WMMA per barrier pair per wave.
// ---------------------------------------------------------------------------
#define TPAD 72  // 64 K elems + 8 pad (keeps 16B alignment, dodges bank conflicts)

__device__ __forceinline__ void gemm_tile_core(const bf16* __restrict__ A,
                                               const bf16* __restrict__ W, int K,
                                               int gm0, int gn0, bf16* Xt, bf16* Wt,
                                               v8f acc[2][4]) {
  const int tid  = threadIdx.x;
  const int lane = tid & 31;
  const int w    = tid >> 5;
  const int wr   = (w & 3) * 32;
  const int wc   = (w >> 2) * 64;
  const int half = lane >> 4;
  const int l15  = lane & 15;
  const int lr   = tid >> 1;        // staging row 0..127
  const int ls   = (tid & 1) * 32;  // k segment (32 elems = 64B)

#pragma unroll
  for (int i = 0; i < 2; ++i)
#pragma unroll
    for (int j = 0; j < 4; ++j)
#pragma unroll
      for (int e = 0; e < 8; ++e) acc[i][j][e] = 0.f;

  for (int kb = 0; kb < K; kb += 64) {
    const bf16* ga = A + (size_t)(gm0 + lr) * K + kb + ls;
    const bf16* gw = W + (size_t)(gn0 + lr) * K + kb + ls;
    bf16* lx = Xt + lr * TPAD + ls;
    bf16* lw = Wt + lr * TPAD + ls;
#pragma unroll
    for (int u = 0; u < 4; ++u) {
      copy16B(ga + u * 8, lx + u * 8);
      copy16B(gw + u * 8, lw + u * 8);
    }
    if (kb + 64 < K) {  // near-scope prefetch of next K tile
      __builtin_prefetch(ga + 64, 0, 3);
      __builtin_prefetch(gw + 64, 0, 3);
    }
    stage_sync();
#pragma unroll
    for (int kk = 0; kk < 2; ++kk) {
      v16bf a0 = load_frag(Xt + (wr + 0 + l15) * TPAD + kk * 32, half);
      v16bf a1 = load_frag(Xt + (wr + 16 + l15) * TPAD + kk * 32, half);
#pragma unroll
      for (int j = 0; j < 4; ++j) {
        v16bf b = load_frag(Wt + (wc + j * 16 + l15) * TPAD + kk * 32, half);
        acc[0][j] = wmma_bf16(a0, b, acc[0][j]);
        acc[1][j] = wmma_bf16(a1, b, acc[1][j]);
      }
    }
    __syncthreads();
  }
}

// ---------------------------------------------------------------------------
// Q projection: Q = x @ Wq^T + b, fused RoPE, write bf16 [B, 32, T, 128]
// ---------------------------------------------------------------------------
__global__ void __launch_bounds__(256) qproj_rope_kernel(
    const bf16* __restrict__ xb, const bf16* __restrict__ wqb,
    const float* __restrict__ qbias, const float* __restrict__ fcos,
    const float* __restrict__ fsin, bf16* __restrict__ qout) {
  __shared__ __align__(16) bf16 smem[2 * 128 * TPAD];
  const int gm0 = blockIdx.y * 128;
  const int gn0 = blockIdx.x * 128;
  v8f acc[2][4];
  gemm_tile_core(xb, wqb, 4096, gm0, gn0, smem, smem + 128 * TPAD, acc);

  const int tid = threadIdx.x, lane = tid & 31, w = tid >> 5;
  const int wr = (w & 3) * 32, wc = (w >> 2) * 64;
  const int half = lane >> 4, l15 = lane & 15;
#pragma unroll
  for (int i = 0; i < 2; ++i)
#pragma unroll
    for (int j = 0; j < 4; ++j) {
      const int n = gn0 + wc + j * 16 + l15;
      const float bias = qbias[n];
      const int p = (n & 127) >> 1;
      const int odd = n & 1;
#pragma unroll
      for (int r = 0; r < 8; ++r) {
        const int m = gm0 + wr + i * 16 + r + 8 * half;
        float v = acc[i][j][r] + bias;
        float partner = __shfl_xor(v, 1, 32);  // pair column n^1
        const int t = m & 2047;
        const float c = fcos[t * 64 + p];
        const float s = fsin[t * 64 + p];
        const float out = odd ? (partner * s + v * c) : (v * c - partner * s);
        const int h = n >> 7, d = n & 127, b = m >> 11;
        qout[(((size_t)(b * 32 + h)) * 2048 + t) * 128 + d] = (bf16)out;
      }
    }
}

// ---------------------------------------------------------------------------
// KV projection: kv = x @ Wkv^T + b. cols [0,1024) -> K (+RoPE) bf16 [B,8,T,128]
// cols [1024,2048) -> V stored transposed bf16 [B,8,128,T]
// ---------------------------------------------------------------------------
__global__ void __launch_bounds__(256) kvproj_rope_kernel(
    const bf16* __restrict__ xb, const bf16* __restrict__ wkvb,
    const float* __restrict__ kvbias, const float* __restrict__ fcos,
    const float* __restrict__ fsin, bf16* __restrict__ kout,
    bf16* __restrict__ vtout) {
  __shared__ __align__(16) bf16 smem[2 * 128 * TPAD];
  const int gm0 = blockIdx.y * 128;
  const int gn0 = blockIdx.x * 128;
  v8f acc[2][4];
  gemm_tile_core(xb, wkvb, 4096, gm0, gn0, smem, smem + 128 * TPAD, acc);

  const int tid = threadIdx.x, lane = tid & 31, w = tid >> 5;
  const int wr = (w & 3) * 32, wc = (w >> 2) * 64;
  const int half = lane >> 4, l15 = lane & 15;
#pragma unroll
  for (int i = 0; i < 2; ++i)
#pragma unroll
    for (int j = 0; j < 4; ++j) {
      const int n = gn0 + wc + j * 16 + l15;
      const float bias = kvbias[n];
#pragma unroll
      for (int r = 0; r < 8; ++r) {
        const int m = gm0 + wr + i * 16 + r + 8 * half;
        float v = acc[i][j][r] + bias;
        float partner = __shfl_xor(v, 1, 32);
        const int t = m & 2047, b = m >> 11;
        if (n < 1024) {  // K head with RoPE
          const int p = (n & 127) >> 1;
          const float c = fcos[t * 64 + p];
          const float s = fsin[t * 64 + p];
          const float out = (n & 1) ? (partner * s + v * c) : (v * c - partner * s);
          const int h = n >> 7, d = n & 127;
          kout[(((size_t)(b * 8 + h)) * 2048 + t) * 128 + d] = (bf16)out;
        } else {  // V, transposed so PV's B operand is contiguous
          const int o2 = n - 1024, h = o2 >> 7, d = o2 & 127;
          vtout[(((size_t)(b * 8 + h)) * 128 + d) * 2048 + t] = (bf16)v;
        }
      }
    }
}

// ---------------------------------------------------------------------------
// Flash attention, causal, GQA (4 q-heads per kv head).
// Block = 128 q rows of one head; 8 waves, each owns a 16-row strip.
// ---------------------------------------------------------------------------
#define QPAD 136
#define KPAD 136
#define VPAD 72
#define PPAD 72

__global__ void __launch_bounds__(256) attn_kernel(const bf16* __restrict__ qm,
                                                   const bf16* __restrict__ km,
                                                   const bf16* __restrict__ vtm,
                                                   bf16* __restrict__ ym) {
  // phase 1: Qb[128][136] (34816B). phase 2: Kt 17408B | Vt 18432B | Pt 18432B
  __shared__ __align__(16) char smem_raw[55296];
  bf16* Qb = (bf16*)smem_raw;
  bf16* Kt = (bf16*)smem_raw;
  bf16* Vt = (bf16*)(smem_raw + 17408);
  bf16* Pt = (bf16*)(smem_raw + 35840);

  const int tid = threadIdx.x, lane = tid & 31, w = tid >> 5;
  const int half = lane >> 4, l15 = lane & 15;
  const int qblk = blockIdx.x;        // 0..15
  const int head = blockIdx.y & 31;
  const int b    = blockIdx.y >> 5;
  const int kvh  = head >> 2;

  const bf16* qbase = qm + (((size_t)(b * 32 + head)) * 2048 + qblk * 128) * 128;
  const bf16* kbase = km + ((size_t)(b * 8 + kvh)) * 2048 * 128;
  const bf16* vbase = vtm + ((size_t)(b * 8 + kvh)) * 128 * 2048;

  {  // stage Q tile 128x128
    const int r = tid >> 1, s = (tid & 1) * 64;
    const bf16* g = qbase + r * 128 + s;
    bf16* l = Qb + r * QPAD + s;
#pragma unroll
    for (int u = 0; u < 8; ++u) copy16B(g + u * 8, l + u * 8);
  }
  stage_sync();
  v16bf qf[4];  // register-resident Q strip (16 rows x 128 d)
#pragma unroll
  for (int kk = 0; kk < 4; ++kk)
    qf[kk] = load_frag(Qb + (w * 16 + l15) * QPAD + kk * 32, half);
  __syncthreads();

  v8f oacc[8];
#pragma unroll
  for (int n = 0; n < 8; ++n)
#pragma unroll
    for (int e = 0; e < 8; ++e) oacc[n][e] = 0.f;
  float mrun[8], srun[8];
#pragma unroll
  for (int r = 0; r < 8; ++r) { mrun[r] = -1e30f; srun[r] = 0.f; }

  const float scale = 0.08838834764831845f;  // 1/sqrt(128)
  const int nlb = 2 * qblk + 2;
  for (int lb = 0; lb < nlb; ++lb) {
    {  // K tile 64x128
      const int r = tid >> 2, s = (tid & 3) * 32;
      const bf16* g = kbase + ((size_t)(lb * 64 + r)) * 128 + s;
      bf16* l = Kt + r * KPAD + s;
#pragma unroll
      for (int u = 0; u < 4; ++u) copy16B(g + u * 8, l + u * 8);
    }
    {  // V^T tile 128x64
      const int d = tid >> 1, s = (tid & 1) * 32;
      const bf16* g = vbase + (size_t)d * 2048 + lb * 64 + s;
      bf16* l = Vt + d * VPAD + s;
#pragma unroll
      for (int u = 0; u < 4; ++u) copy16B(g + u * 8, l + u * 8);
    }
    stage_sync();

    // S strip (16 x 64) = Q K^T
    v8f sacc[4];
#pragma unroll
    for (int j = 0; j < 4; ++j)
#pragma unroll
      for (int e = 0; e < 8; ++e) sacc[j][e] = 0.f;
#pragma unroll
    for (int kk = 0; kk < 4; ++kk)
#pragma unroll
      for (int j = 0; j < 4; ++j) {
        v16bf bk = load_frag(Kt + (j * 16 + l15) * KPAD + kk * 32, half);
        sacc[j] = wmma_bf16(qf[kk], bk, sacc[j]);
      }

    // scale + causal mask + online softmax update
    float pv[4][8], alpha[8];
    const int qrow0 = qblk * 128 + w * 16 + 8 * half;
    const int lcol0 = lb * 64 + l15;
#pragma unroll
    for (int r = 0; r < 8; ++r) {
      const int q = qrow0 + r;
      float mx = -1e30f;
#pragma unroll
      for (int j = 0; j < 4; ++j) {
        float s = sacc[j][r] * scale;
        if (lcol0 + j * 16 > q) s = -1e30f;
        pv[j][r] = s;
        mx = fmaxf(mx, s);
      }
      for (int off = 1; off < 16; off <<= 1)  // reduce within 16-lane half
        mx = fmaxf(mx, __shfl_xor(mx, off, 32));
      const float mnew = fmaxf(mrun[r], mx);
      alpha[r] = __expf(mrun[r] - mnew);
      float rsum = 0.f;
#pragma unroll
      for (int j = 0; j < 4; ++j) {
        const float p = __expf(pv[j][r] - mnew);
        pv[j][r] = p;
        rsum += p;
      }
      for (int off = 1; off < 16; off <<= 1) rsum += __shfl_xor(rsum, off, 32);
      srun[r] = srun[r] * alpha[r] + rsum;
      mrun[r] = mnew;
    }
#pragma unroll
    for (int n = 0; n < 8; ++n)
#pragma unroll
      for (int r = 0; r < 8; ++r) oacc[n][r] *= alpha[r];

    // bounce P through per-wave LDS (D-layout -> A-layout), then O += P V
    bf16* Pw = Pt + w * 16 * PPAD;
#pragma unroll
    for (int j = 0; j < 4; ++j)
#pragma unroll
      for (int r = 0; r < 8; ++r)
        Pw[(r + 8 * half) * PPAD + j * 16 + l15] = (bf16)pv[j][r];
#pragma unroll
    for (int kk = 0; kk < 2; ++kk) {
      v16bf ap = load_frag(Pw + l15 * PPAD + kk * 32, half);
#pragma unroll
      for (int n = 0; n < 8; ++n) {
        v16bf bv = load_frag(Vt + (n * 16 + l15) * VPAD + kk * 32, half);
        oacc[n] = wmma_bf16(ap, bv, oacc[n]);
      }
    }
    __syncthreads();
  }

  // normalize and write y[b, t, head*128 + d] (bf16)
  const int qrow0 = qblk * 128 + w * 16 + 8 * half;
  float inv[8];
#pragma unroll
  for (int r = 0; r < 8; ++r) inv[r] = 1.0f / srun[r];
#pragma unroll
  for (int n = 0; n < 8; ++n) {
    const int d = n * 16 + l15;
#pragma unroll
    for (int r = 0; r < 8; ++r) {
      const int t = qrow0 + r;
      ym[((size_t)(b * 2048 + t)) * 4096 + head * 128 + d] =
          (bf16)(oacc[n][r] * inv[r]);
    }
  }
}

// ---------------------------------------------------------------------------
// Output projection: out = y @ Wp^T + b  (fp32 result)
// ---------------------------------------------------------------------------
__global__ void __launch_bounds__(256) outproj_kernel(
    const bf16* __restrict__ yb, const bf16* __restrict__ pwb,
    const float* __restrict__ pbias, float* __restrict__ out) {
  __shared__ __align__(16) bf16 smem[2 * 128 * TPAD];
  const int gm0 = blockIdx.y * 128;
  const int gn0 = blockIdx.x * 128;
  v8f acc[2][4];
  gemm_tile_core(yb, pwb, 4096, gm0, gn0, smem, smem + 128 * TPAD, acc);

  const int tid = threadIdx.x, lane = tid & 31, w = tid >> 5;
  const int wr = (w & 3) * 32, wc = (w >> 2) * 64;
  const int half = lane >> 4, l15 = lane & 15;
#pragma unroll
  for (int i = 0; i < 2; ++i)
#pragma unroll
    for (int j = 0; j < 4; ++j) {
      const int n = gn0 + wc + j * 16 + l15;
      const float bias = pbias[n];
#pragma unroll
      for (int r = 0; r < 8; ++r) {
        const int m = gm0 + wr + i * 16 + r + 8 * half;
        out[(size_t)m * 4096 + n] = acc[i][j][r] + bias;
      }
    }
}

// ---------------------------------------------------------------------------
extern "C" void kernel_launch(void* const* d_in, const int* in_sizes, int n_in,
                              void* d_out, int out_size, void* d_ws, size_t ws_size,
                              hipStream_t stream) {
  const float* x      = (const float*)d_in[0];
  const float* fcos   = (const float*)d_in[1];
  const float* fsin   = (const float*)d_in[2];
  const float* wq     = (const float*)d_in[3];
  const float* qbias  = (const float*)d_in[4];
  const float* wkv    = (const float*)d_in[5];
  const float* kvbias = (const float*)d_in[6];
  const float* pw     = (const float*)d_in[7];
  const float* pbias  = (const float*)d_in[8];
  float* out = (float*)d_out;

  // workspace carve-out (all 256B aligned)
  char* ws = (char*)d_ws;
  size_t off = 0;
  auto alloc = [&](size_t bytes) {
    void* p = ws + off;
    off += (bytes + 255) & ~(size_t)255;
    return p;
  };
  const int M = 4096;  // B*T
  bf16* xb   = (bf16*)alloc((size_t)M * 4096 * 2);    // x bf16
  bf16* wqb  = (bf16*)alloc((size_t)4096 * 4096 * 2); // Wq bf16
  bf16* wkvb = (bf16*)alloc((size_t)2048 * 4096 * 2); // Wkv bf16
  bf16* pwb  = (bf16*)alloc((size_t)4096 * 4096 * 2); // Wproj bf16
  bf16* qbf  = (bf16*)alloc((size_t)2 * 32 * 2048 * 128 * 2); // Q roped [B,32,T,D]
  bf16* kbf  = (bf16*)alloc((size_t)2 * 8 * 2048 * 128 * 2);  // K roped [B,8,T,D]
  bf16* vtbf = (bf16*)alloc((size_t)2 * 8 * 2048 * 128 * 2);  // V^T     [B,8,D,T]
  bf16* ybf  = (bf16*)alloc((size_t)M * 4096 * 2);            // attn out [B*T, 4096]

  cvt_f32_bf16<<<4096, 256, 0, stream>>>(x, xb, M * 4096);
  cvt_f32_bf16<<<4096, 256, 0, stream>>>(wq, wqb, 4096 * 4096);
  cvt_f32_bf16<<<4096, 256, 0, stream>>>(wkv, wkvb, 2048 * 4096);
  cvt_f32_bf16<<<4096, 256, 0, stream>>>(pw, pwb, 4096 * 4096);

  qproj_rope_kernel<<<dim3(32, 32), 256, 0, stream>>>(xb, wqb, qbias, fcos, fsin,
                                                      qbf);
  kvproj_rope_kernel<<<dim3(16, 32), 256, 0, stream>>>(xb, wkvb, kvbias, fcos,
                                                       fsin, kbf, vtbf);
  attn_kernel<<<dim3(16, 64), 256, 0, stream>>>(qbf, kbf, vtbf, ybf);
  outproj_kernel<<<dim3(32, 32), 256, 0, stream>>>(ybf, pwb, pbias, out);
}
